// MeshGNN_79534204387339
// MI455X (gfx1250) — matrix operand order repaired and checked by reading, not compiled, not run on hardware
//
#include <hip/hip_runtime.h>

#define HIDDEN 128

typedef __attribute__((ext_vector_type(16))) __bf16 v16bf;
typedef __attribute__((ext_vector_type(8)))  float  v8f;

union Frag { v16bf v; uint4 q[2]; };

__device__ __forceinline__ unsigned short f32_to_bf16_rne(float f) {
    unsigned int u = __builtin_bit_cast(unsigned int, f);
    u += 0x7FFFu + ((u >> 16) & 1u);   // round-to-nearest-even
    return (unsigned short)(u >> 16);
}

__device__ __forceinline__ float bf16_to_f32(unsigned short u) {
    return __builtin_bit_cast(float, (unsigned int)u << 16);
}

// ---------------- degree / normalization ----------------

__global__ void k_init_deg(float* deg, int nN) {
    int i = blockIdx.x * blockDim.x + threadIdx.x;
    if (i < nN) deg[i] = 1.0f;                       // self-loop contributes 1
}

__global__ void k_count_deg(const long long* __restrict__ ei, float* deg, int nE) {
    int e = blockIdx.x * blockDim.x + threadIdx.x;
    if (e < nE) atomicAdd(&deg[(int)ei[(size_t)nE + e]], 1.0f);   // dst row
}

__global__ void k_dinv(const float* __restrict__ deg, float* dinv, int nN) {
    int i = blockIdx.x * blockDim.x + threadIdx.x;
    if (i < nN) {
        float d = deg[i];
        dinv[i] = (d > 0.0f) ? rsqrtf(d) : 0.0f;
    }
}

// ------------- weight transpose + conversion: Wt[out][in] = bf16(W[in][out]) -------------

__global__ void k_convWt(const float* __restrict__ W, unsigned short* __restrict__ Wt) {
    int t = blockIdx.x * blockDim.x + threadIdx.x;
    if (t >= HIDDEN * HIDDEN) return;
    int col = t >> 7;            // out channel
    int k   = t & (HIDDEN - 1);  // in channel
    Wt[t] = f32_to_bf16_rne(W[k * HIDDEN + col]);
}

// ---------------- encoder dense: T = x(N,6) @ W_enc(6,128) ----------------

__global__ void k_dense_enc(const float* __restrict__ x, const float* __restrict__ W,
                            float* __restrict__ T, int nN) {
    int t = blockIdx.x * blockDim.x + threadIdx.x;
    if (t >= nN * HIDDEN) return;
    int n = t >> 7, c = t & (HIDDEN - 1);
    const float* xr = x + (size_t)n * 6;
    float s = 0.0f;
#pragma unroll
    for (int k = 0; k < 6; ++k) s += xr[k] * W[k * HIDDEN + c];
    T[t] = s;
}

// ------------- hidden dense via WMMA: T(f32) = Hbf(bf16) @ Wt^T(bf16) -------------
// One wave = 32 nodes x 128 channels (2 M-tiles, 16 accumulators, 64 WMMAs).
// Wt (transposed weights) staged in LDS; B fragments double-buffered ds_load_b128 pairs.

__global__ void __launch_bounds__(128)
k_dense_hid_wmma(const unsigned short* __restrict__ Hbf,
                 const unsigned short* __restrict__ Wt,
                 float* __restrict__ T, int nN) {
    __shared__ uint4 sW4[2048];                            // 32KB transposed bf16 weights
    const uint4* gW = (const uint4*)Wt;
    for (int i = threadIdx.x; i < 2048; i += blockDim.x) sW4[i] = gW[i];
    __syncthreads();
    const char* sWb = (const char*)sW4;

    const int wave = threadIdx.x >> 5;
    const int lane = threadIdx.x & 31;
    const int nodeBase = (blockIdx.x * 4 + wave) * 32;     // 32 nodes per wave
    if (nodeBase >= nN) return;                            // wave-uniform guard (EXEC all-1s)

    const int m    = lane & 15;
    const int half = lane >> 4;

    v8f acc0[8] = {};
    v8f acc1[8] = {};

    // A rows: lane m handles node rows (nodeBase+m) and (nodeBase+16+m)
    const char* a0 = (const char*)(Hbf + (size_t)(nodeBase + m) * HIDDEN);
    const char* a1 = a0 + (size_t)16 * HIDDEN * sizeof(unsigned short);
    // B base: column (nt*16+m), K run starts at half*16; per-lane base + immediate offsets
    const char* bbase = sWb + ((size_t)m * HIDDEN + half * 16) * sizeof(unsigned short);

    // byte offset of the B fragment for (kc, nt)
    auto boff = [](int kc, int nt) { return nt * (16 * HIDDEN * 2) + kc * 64; };

    Frag Bbuf[2];
    Bbuf[0].q[0] = *(const uint4*)(bbase + boff(0, 0));    // prologue: prefetch first B
    Bbuf[0].q[1] = *(const uint4*)(bbase + boff(0, 0) + 16);

#pragma unroll
    for (int kc = 0; kc < 4; ++kc) {                       // K = 128 in chunks of 32
        const int aoff = kc * 64 + half * 32;              // bytes: (kc*32 + half*16) bf16
        Frag A0, A1;
        A0.q[0] = *(const uint4*)(a0 + aoff);
        A0.q[1] = *(const uint4*)(a0 + aoff + 16);         // +8 bf16 (K +8)
        A1.q[0] = *(const uint4*)(a1 + aoff);
        A1.q[1] = *(const uint4*)(a1 + aoff + 16);

#pragma unroll
        for (int nt = 0; nt < 8; ++nt) {                   // 8 x 16 output channels
            const int cur = (kc * 8 + nt) & 1;
            // prefetch next B fragment into the alternate buffer (overlaps with WMMAs below)
            if (!(kc == 3 && nt == 7)) {
                const int nkc = (nt == 7) ? kc + 1 : kc;
                const int nnt = (nt == 7) ? 0 : nt + 1;
                Bbuf[cur ^ 1].q[0] = *(const uint4*)(bbase + boff(nkc, nnt));
                Bbuf[cur ^ 1].q[1] = *(const uint4*)(bbase + boff(nkc, nnt) + 16);
            }
            acc0[nt] = __builtin_amdgcn_wmma_f32_16x16x32_bf16(
                false, A0.v, false, Bbuf[cur].v, (short)0, acc0[nt], false, false);
            acc1[nt] = __builtin_amdgcn_wmma_f32_16x16x32_bf16(
                false, A1.v, false, Bbuf[cur].v, (short)0, acc1[nt], false, false);
        }
    }

    // store D: row = r + 8*half (+16 for tile 1), col = nt*16 + m
#pragma unroll
    for (int nt = 0; nt < 8; ++nt) {
#pragma unroll
        for (int r = 0; r < 8; ++r) {
            const size_t row0 = (size_t)(nodeBase + r + 8 * half);
            T[row0 * HIDDEN + nt * 16 + m]        = acc0[nt][r];
            T[(row0 + 16) * HIDDEN + nt * 16 + m] = acc1[nt][r];
        }
    }
}

// ---------------- decoder dense: T3 = Hbf(N,128 bf16) @ W_dec(128,3) ----------------

__global__ void k_dense_dec(const unsigned short* __restrict__ Hbf, const float* __restrict__ W,
                            float* __restrict__ T3, int nN) {
    int t = blockIdx.x * blockDim.x + threadIdx.x;
    if (t >= nN * 3) return;
    int n = t / 3, c = t - n * 3;
    const unsigned short* hr = Hbf + (size_t)n * HIDDEN;
    float s = 0.0f;
#pragma unroll 8
    for (int k = 0; k < HIDDEN; ++k) s += bf16_to_f32(hr[k]) * W[k * 3 + c];
    T3[t] = s;
}

// ---------------- aggregation: self-loop init + edge scatter ----------------

__global__ void k_agg_self128(const float* __restrict__ T, const float* __restrict__ dinv,
                              float* __restrict__ out, int nN) {
    int t = blockIdx.x * blockDim.x + threadIdx.x;
    if (t >= nN * HIDDEN) return;
    int n = t >> 7;
    float di = dinv[n];
    out[t] = T[t] * di * di;
}

__global__ void k_agg_edges128(const long long* __restrict__ ei, const float* __restrict__ T,
                               const float* __restrict__ dinv, float* __restrict__ out, int nE) {
    int t = blockIdx.x * blockDim.x + threadIdx.x;
    int e = t >> 5;
    if (e >= nE) return;
    int q   = (t & 31) << 2;                               // 4 channels per lane
    int src = (int)ei[e];
    int dst = (int)ei[(size_t)nE + e];
    float nrm = dinv[src] * dinv[dst];
    float4 msg = *(const float4*)(T + (size_t)src * HIDDEN + q);
    float* o = out + (size_t)dst * HIDDEN + q;
    atomicAdd(o + 0, msg.x * nrm);
    atomicAdd(o + 1, msg.y * nrm);
    atomicAdd(o + 2, msg.z * nrm);
    atomicAdd(o + 3, msg.w * nrm);
}

__global__ void k_agg_self3(const float* __restrict__ T3, const float* __restrict__ dinv,
                            float* __restrict__ out, int nN) {
    int t = blockIdx.x * blockDim.x + threadIdx.x;
    if (t >= nN * 3) return;
    int n = t / 3;
    float di = dinv[n];
    out[t] = T3[t] * di * di;
}

__global__ void k_agg_edges3(const long long* __restrict__ ei, const float* __restrict__ T3,
                             const float* __restrict__ dinv, float* __restrict__ out, int nE) {
    int e = blockIdx.x * blockDim.x + threadIdx.x;
    if (e >= nE) return;
    int src = (int)ei[e];
    int dst = (int)ei[(size_t)nE + e];
    float nrm = dinv[src] * dinv[dst];
#pragma unroll
    for (int c = 0; c < 3; ++c)
        atomicAdd(&out[(size_t)dst * 3 + c], T3[(size_t)src * 3 + c] * nrm);
}

// ---------------- bias + relu -> bf16 activations ----------------

__global__ void k_bias_relu_bf16(const float* __restrict__ A, const float* __restrict__ b,
                                 unsigned short* __restrict__ O, int total) {
    int t = blockIdx.x * blockDim.x + threadIdx.x;
    if (t >= total) return;
    float v = A[t] + b[t & (HIDDEN - 1)];
    O[t] = f32_to_bf16_rne(v > 0.0f ? v : 0.0f);
}

__global__ void k_bias3(float* __restrict__ A, const float* __restrict__ b, int total) {
    int t = blockIdx.x * blockDim.x + threadIdx.x;
    if (t >= total) return;
    int c = t - (t / 3) * 3;
    A[t] = A[t] + b[c];
}

// ---------------- launcher ----------------

extern "C" void kernel_launch(void* const* d_in, const int* in_sizes, int n_in,
                              void* d_out, int out_size, void* d_ws, size_t ws_size,
                              hipStream_t stream) {
    const float*     x     = (const float*)d_in[0];
    const long long* ei    = (const long long*)d_in[1];   // int64 [2, E]
    const float*     W_enc = (const float*)d_in[2];
    const float*     b_enc = (const float*)d_in[3];
    const float*     W_hid = (const float*)d_in[4];
    const float*     b_hid = (const float*)d_in[5];
    const float*     W_dec = (const float*)d_in[6];
    const float*     b_dec = (const float*)d_in[7];

    const int nN = in_sizes[0] / 6;
    const int nE = in_sizes[1] / 2;
    float* out = (float*)d_out;

    // workspace layout
    char*  ws   = (char*)d_ws;
    size_t szH  = (size_t)nN * HIDDEN * sizeof(float);          // N x 128 f32
    float*          hbuf = (float*)ws;                          // aggregation target (f32)
    float*          tbuf = (float*)(ws + szH);                  // dense output (f32); also N x 3
    unsigned short* hbf  = (unsigned short*)(ws + 2 * szH);     // activations (bf16)
    float*          deg  = (float*)(ws + 2 * szH + szH / 2);
    float*          dinv = (float*)(ws + 2 * szH + szH / 2 + (size_t)nN * 4);
    unsigned short* Wt   = (unsigned short*)(ws + 2 * szH + szH / 2 + (size_t)nN * 8);

    const int B = 256;
    const int gN    = (nN + B - 1) / B;
    const int gE    = (nE + B - 1) / B;
    const int gNC   = (nN * HIDDEN + B - 1) / B;
    const int gE32  = (int)(((size_t)nE * 32 + B - 1) / B);
    const int gN3   = (nN * 3 + B - 1) / B;
    const int waves = (nN + 31) / 32;                           // 32 nodes per wave
    const int gWmma = (waves + 3) / 4;                          // 4 waves per block

    // --- gcn_norm ---
    k_init_deg <<<gN, B, 0, stream>>>(deg, nN);
    k_count_deg<<<gE, B, 0, stream>>>(ei, deg, nE);
    k_dinv     <<<gN, B, 0, stream>>>(deg, dinv, nN);
    k_convWt   <<<(HIDDEN * HIDDEN + B - 1) / B, B, 0, stream>>>(W_hid, Wt);

    // --- encoder conv + relu ---
    k_dense_enc    <<<gNC,  B, 0, stream>>>(x, W_enc, tbuf, nN);
    k_agg_self128  <<<gNC,  B, 0, stream>>>(tbuf, dinv, hbuf, nN);
    k_agg_edges128 <<<gE32, B, 0, stream>>>(ei, tbuf, dinv, hbuf, nE);
    k_bias_relu_bf16<<<gNC, B, 0, stream>>>(hbuf, b_enc, hbf, nN * HIDDEN);

    // --- 4 hidden convs + relu (WMMA GEMM on bf16 activations) ---
    for (int l = 0; l < 4; ++l) {
        k_dense_hid_wmma<<<gWmma, 128, 0, stream>>>(hbf, Wt, tbuf, nN);
        k_agg_self128   <<<gNC,  B, 0, stream>>>(tbuf, dinv, hbuf, nN);
        k_agg_edges128  <<<gE32, B, 0, stream>>>(ei, tbuf, dinv, hbuf, nE);
        k_bias_relu_bf16<<<gNC,  B, 0, stream>>>(hbuf, b_hid, hbf, nN * HIDDEN);
    }

    // --- decoder conv (no relu) -> d_out ---
    k_dense_dec <<<gN3, B, 0, stream>>>(hbf, W_dec, tbuf, nN);
    k_agg_self3 <<<gN3, B, 0, stream>>>(tbuf, dinv, out, nN);
    k_agg_edges3<<<gE,  B, 0, stream>>>(ei, tbuf, dinv, out, nE);
    k_bias3     <<<gN3, B, 0, stream>>>(out, b_dec, nN * 3);
}
